// ElementalModesMessagePassingNeuralNetwork_35527969473041
// MI455X (gfx1250) — matrix operand order repaired and checked by reading, not compiled, not run on hardware
//
#include <hip/hip_runtime.h>

// ---------------- problem constants ----------------
#define NA    10000      // atoms  (divisible by 16 -> 625 row tiles)
#define NE    320000     // edges  (divisible by 16 -> 20000 row tiles)
#define FDIM  128
#define KRBF  64
#define NOUTC 2001
#define NB    5
#define MT_A  (NA / 16)  // 625
#define NTOT  (NA * NOUTC)

// LDS row strides (elements) with 16B padding to avoid bank conflicts
#define LDA_S  136   // 128 + 8 bf16
#define LDR_S  72    // 64 + 8 bf16

static_assert(NA % 16 == 0 && NE % 16 == 0, "tile divisibility");

typedef __attribute__((ext_vector_type(16))) __bf16 v16bf;
typedef __attribute__((ext_vector_type(8)))  __bf16 v8bf;
typedef __attribute__((ext_vector_type(8)))  float  v8f;
typedef __attribute__((ext_vector_type(4)))  unsigned int v4u;
typedef __attribute__((ext_vector_type(8)))  int v8i;
typedef __attribute__((ext_vector_type(4)))  int v4i;

// shifted softplus: softplus(x) - log(2)
__device__ __forceinline__ float sspf(float x) {
  return fmaxf(x, 0.f) + log1pf(__expf(-fabsf(x))) - 0.69314718055994531f;
}

__device__ __forceinline__ v16bf cat8(v8bf lo, v8bf hi) {
  v16bf r;
#pragma unroll
  for (int i = 0; i < 8; ++i) { r[i] = lo[i]; r[i + 8] = hi[i]; }
  return r;
}

// A fragment (16x32 bf16, MxK) from a row-major bf16 tile (global or LDS).
// Lane layout (ISA 7.12.2): lanes 0-15 row=lane, K = {0..7, 16..23};
// lanes 16-31 row=lane-16, K = {8..15, 24..31}.
__device__ __forceinline__ v16bf load_afrag(const __bf16* A, int ldA,
                                            int row, int kbase, int lane) {
  const __bf16* p = A + (size_t)row * ldA + kbase + ((lane & 16) ? 8 : 0);
  v8bf lo = *(const v8bf*)p;
  v8bf hi = *(const v8bf*)(p + 16);
  return cat8(lo, hi);
}

// B fragment (32x16 bf16, KxN) from pre-packed fragment-major buffer:
// frag*512 + lane*16 + e, where n = lane&15, k = (lane>>4)*8 + (e&7) + (e>>3)*16.
__device__ __forceinline__ v16bf load_bfrag(const __bf16* __restrict__ Bp,
                                            int frag, int lane) {
  const __bf16* p = Bp + ((size_t)frag * 32 + lane) * 16;
  v8bf lo = *(const v8bf*)p;
  v8bf hi = *(const v8bf*)(p + 8);
  return cat8(lo, hi);
}

// ---------------- TDM: load one 16x64 bf16 tile into LDS with padding ----
// D# per CDNA5 ISA: group0 {count|lds|gaddr|type=2},
// group1 {data_size=2B, pad every 128B by 16B, dims 64x16, stride 64}.
__device__ __forceinline__ void tdm_load_rbf_tile(unsigned ldsOff,
                                                  const __bf16* gsrc) {
  unsigned long long ga = (unsigned long long)(uintptr_t)gsrc;
  v4u g0;
  g0[0] = 1u;                                   // count=1, user mode
  g0[1] = ldsOff;                               // lds_addr (bytes)
  g0[2] = (unsigned)(ga & 0xFFFFFFFFu);         // global_addr[31:0]
  g0[3] = (unsigned)((ga >> 32) & 0x1FFFFFFu) | (2u << 30);  // addr[56:32]|type=2
  v8i g1;
  g1[0] = (1 << 16)      // data_size = 2 bytes
        | (1 << 20)      // pad_enable
        | (4 << 22)      // pad_interval: 128 bytes
        | (3 << 25);     // pad_amount: 4 DWORDs = 16 bytes
  g1[1] = (KRBF << 16);  // tensor_dim0[15:0]
  g1[2] = (16 << 16);    // tensor_dim0[31:16]=0 | tensor_dim1[15:0]=16
  g1[3] = (KRBF << 16);  // tensor_dim1[31:16]=0 | tile_dim0=64
  g1[4] = 16;            // tile_dim1=16 | tile_dim2=0
  g1[5] = KRBF;          // tensor_dim0_stride[31:0]=64
  g1[6] = 0;
  g1[7] = 0;
  v4i z4 = {};
  v8i z8 = {};
  __builtin_amdgcn_tensor_load_to_lds(g0, g1, z4, z4, z8, 0);
  __builtin_amdgcn_s_wait_tensorcnt(0);
}

// ---------------- weight packing ----------------
__global__ void k_pack(const float* __restrict__ W, __bf16* __restrict__ out,
                       int K, int N, int Ntiles, int total) {
  int idx = blockIdx.x * 256 + threadIdx.x;
  if (idx >= total) return;
  int frag = idx >> 9;
  int lane = (idx >> 4) & 31;
  int e    = idx & 15;
  int kt = frag / Ntiles, nt = frag % Ntiles;
  int n = nt * 16 + (lane & 15);
  int k = kt * 32 + (lane >> 4) * 8 + (e & 7) + (e >> 3) * 16;
  float v = (k < K && n < N) ? W[(size_t)k * N + n] : 0.f;
  out[idx] = (__bf16)v;
}

// ---------------- elementwise ----------------
__global__ void k_zero(float* p, int n) {
  int i = blockIdx.x * 256 + threadIdx.x;
  if (i < n) p[i] = 0.f;
}

__global__ void k_ssp_bf16(const float* __restrict__ x, __bf16* __restrict__ y, int n) {
  int i = blockIdx.x * 256 + threadIdx.x;
  if (i < n) y[i] = (__bf16)sspf(x[i]);
}

__global__ void k_pad_f(const float* __restrict__ f, __bf16* __restrict__ out) {
  int i = blockIdx.x * 256 + threadIdx.x;           // NA*32 threads
  if (i >= NA * 32) return;
  int a = i >> 5, c = i & 31;
  out[i] = (c < 8) ? (__bf16)f[a * 8 + c] : (__bf16)0.f;
}

// RBF expansion -> bf16 [NE, 64]
__global__ void k_rbf(const float* __restrict__ R, const float* __restrict__ offs,
                      const int* __restrict__ ii, const int* __restrict__ jj,
                      __bf16* __restrict__ out) {
  int t = blockIdx.x * 256 + threadIdx.x;
  if (t >= NE * KRBF) return;
  int e = t >> 6, k = t & 63;
  int i = ii[e], j = jj[e];
  float dx = R[3 * i + 0] - (R[3 * j + 0] + offs[3 * e + 0]);
  float dy = R[3 * i + 1] - (R[3 * j + 1] + offs[3 * e + 1]);
  float dz = R[3 * i + 2] - (R[3 * j + 2] + offs[3 * e + 2]);
  float D = sqrtf(fmaxf(dx * dx + dy * dy + dz * dz, 0.f));
  float p = D * 0.125f;
  float p2 = p * p, p3 = p2 * p;
  float phi = (D < 8.f) ? (1.f - 6.f * p3 * p2 + 15.f * p3 * p - 10.f * p3) : 0.f;
  const float c0 = 3.35462627902512e-4f;                  // exp(-8)
  float center = c0 + (1.f - c0) * ((float)k * (1.f / 63.f));
  const float wspan = 0.03125f * (1.f - c0);              // (2/K)*(1-exp(-8))
  const float width = 1.f / (wspan * wspan);
  float d = __expf(-D) - center;
  out[t] = (__bf16)(phi * __expf(-width * d * d));
}

__global__ void k_finalize(float* __restrict__ out, const float* __restrict__ nh) {
  int i = blockIdx.x * 256 + threadIdx.x;
  if (i < NTOT) out[i] = fmaxf(out[i], 0.f);
  if (i == 0) out[NTOT] = nh[0] * (1.0f / (float)NTOT);
}

// ---------------- generic WMMA GEMM with epilogues ----------------
enum { EP_STORE = 0, EP_SSP = 1, EP_ADD = 2, EP_UX = 3, EP_OUT = 4 };

template <int MODE>
__global__ __launch_bounds__(256) void k_gemm(
    const __bf16* __restrict__ A, int ldA, int Ktiles,
    const __bf16* __restrict__ Bp, int Ntiles,
    const float* __restrict__ bias,
    float* __restrict__ Yf, __bf16* __restrict__ Ybf, int ldY, int Nreal,
    const float* __restrict__ Xbase, const float* __restrict__ u,
    float* __restrict__ nh, float* __restrict__ prevO, int addOut) {
  __shared__ __bf16 sA[16 * LDA_S];   // staged A tile (shared by all 8 waves)
  const int t = threadIdx.x;
  const int mt = blockIdx.x;
  // cooperative stage: 16 rows x Ktiles*32 bf16, 8 bf16 (16B) per thread
  const int chunksPerRow = Ktiles * 4;
  if (t < 16 * chunksPerRow) {
    int row = t / chunksPerRow;
    int cc = (t % chunksPerRow) * 8;
    *(v8bf*)&sA[row * LDA_S + cc] =
        *(const v8bf*)&A[(size_t)(mt * 16 + row) * ldA + cc];
  }
  __syncthreads();

  const int lane = t & 31;
  const int nt = blockIdx.y * 8 + (t >> 5);
  if (nt >= Ntiles) return;
  v8f acc = {};
  for (int kt = 0; kt < Ktiles; ++kt) {
    v16bf a = load_afrag(sA, LDA_S, lane & 15, kt * 32, lane);
    v16bf b = load_bfrag(Bp, kt * Ntiles + nt, lane);
    acc = __builtin_amdgcn_wmma_f32_16x16x32_bf16(false, a, false, b,
                                                  (short)0, acc, false, false);
  }
  const int n = nt * 16 + (lane & 15);
  const bool nok = (n < Nreal);
  const float bv = (bias && nok) ? bias[n] : 0.f;
  const float uv = (MODE == EP_UX && nok) ? u[n] : 0.f;
  const int mofs = (lane & 16) ? 8 : 0;
  float nhpart = 0.f;
#pragma unroll
  for (int r = 0; r < 8; ++r) {
    const int mm = mt * 16 + r + mofs;
    const size_t idx = (size_t)mm * ldY + n;
    const float v = acc[r] + bv;
    if (nok) {
      if (MODE == EP_STORE) {
        Yf[idx] = v;
      } else if (MODE == EP_SSP) {
        float s = sspf(v);
        if (Yf)  Yf[idx] = s;
        if (Ybf) Ybf[idx] = (__bf16)s;
      } else if (MODE == EP_ADD) {
        Yf[idx] = Xbase[idx] + v;
      } else if (MODE == EP_UX) {
        Yf[idx] = uv * Xbase[idx] + v;
      } else {  // EP_OUT: accumulate outputs, nhloss vs previous block's o
        Yf[idx] += v;
        if (addOut) {
          float pv = prevO[idx];
          float v2 = v * v;
          nhpart += v2 / (v2 + pv * pv + 1e-7f);
        }
        prevO[idx] = v;
      }
    }
  }
  if (MODE == EP_OUT && addOut) {
#pragma unroll
    for (int off = 16; off > 0; off >>= 1) nhpart += __shfl_down(nhpart, off, 32);
    if (lane == 0) atomicAdd(nh, nhpart);
  }
}

// ---------------- fused edge message + segment-sum ----------------
// m[idx_i[e], :] += (rbf[e] @ Wg)[:] * ssp((xa[idx_j[e]] @ Wj + bj))[:]
// rbf tile fetched by the Tensor Data Mover (with LDS padding); gathered
// xa rows staged cooperatively so each row is read once per workgroup.
__global__ __launch_bounds__(256) void k_edge(
    const __bf16* __restrict__ rbf, const __bf16* __restrict__ xa,
    const int* __restrict__ ii, const int* __restrict__ jj,
    const __bf16* __restrict__ WgP, const __bf16* __restrict__ WjP,
    const float* __restrict__ bj, float* __restrict__ mOut) {
  __shared__ __bf16 sRbf[16 * LDR_S];
  __shared__ __bf16 sXa[16 * LDA_S];
  __shared__ int sIdx[16];
  const int t = threadIdx.x;
  const int et = blockIdx.x;

  if (t == 0) {
    // warm GL2 for the next tile's rbf rows
    __builtin_prefetch(rbf + (size_t)(et + 1) * 16 * KRBF, 0, 0);
  }
  if (t < 32) {
    // wave 0 drives the TDM; pad_interval=128B/pad_amount=16B produces the
    // bank-conflict-free LDR_S row stride directly in LDS.
    tdm_load_rbf_tile((unsigned)(uintptr_t)&sRbf[0],
                      rbf + (size_t)et * 16 * KRBF);
  }
  {  // gather 16 rows of xa (16 x 128 bf16), 16B per thread
    int row = t >> 4;
    int cc = (t & 15) * 8;
    int jr = jj[et * 16 + row];
    *(v8bf*)&sXa[row * LDA_S + cc] = *(const v8bf*)&xa[(size_t)jr * FDIM + cc];
  }
  if (t < 16) sIdx[t] = ii[et * 16 + t];
  __syncthreads();

  const int lane = t & 31;
  const int nt = t >> 5;  // 8 waves -> 8 column tiles of F=128
  v8f g = {};
#pragma unroll
  for (int kt = 0; kt < 2; ++kt) {
    v16bf a = load_afrag(sRbf, LDR_S, lane & 15, kt * 32, lane);
    v16bf b = load_bfrag(WgP, kt * 8 + nt, lane);
    g = __builtin_amdgcn_wmma_f32_16x16x32_bf16(false, a, false, b, (short)0, g,
                                                false, false);
  }
  v8f jm = {};
#pragma unroll
  for (int kt = 0; kt < 4; ++kt) {
    v16bf a = load_afrag(sXa, LDA_S, lane & 15, kt * 32, lane);
    v16bf b = load_bfrag(WjP, kt * 8 + nt, lane);
    jm = __builtin_amdgcn_wmma_f32_16x16x32_bf16(false, a, false, b, (short)0, jm,
                                                 false, false);
  }
  const int n = nt * 16 + (lane & 15);
  const float bv = bj[n];
  const int mofs = (lane & 16) ? 8 : 0;
#pragma unroll
  for (int r = 0; r < 8; ++r) {
    const float val = g[r] * sspf(jm[r] + bv);
    atomicAdd(&mOut[(size_t)sIdx[r + mofs] * FDIM + n], val);
  }
}

// ---------------- host side ----------------
static inline int cdiv_i(int a, int b) { return (a + b - 1) / b; }

template <int MODE>
static void launch_gemm(hipStream_t s, int Mtiles,
                        const __bf16* A, int ldA, int Ktiles,
                        const __bf16* Bp, int Ntiles, const float* bias,
                        float* Yf, __bf16* Ybf, int ldY, int Nreal,
                        const float* Xbase, const float* u,
                        float* nh, float* prevO, int addOut) {
  dim3 g(Mtiles, cdiv_i(Ntiles, 8));
  k_gemm<MODE><<<g, 256, 0, s>>>(A, ldA, Ktiles, Bp, Ntiles, bias, Yf, Ybf, ldY,
                                 Nreal, Xbase, u, nh, prevO, addOut);
}

// Parameter-leaf index map. Supports both plausible harness flattenings:
//   insertion order: R,f,offsets,<params leaves in dict insertion order>,idx_i,idx_j
//   jax sorted keys: R,f,idx_i,idx_j,offsets,<params leaves with sorted keys>
struct PIdx {
  int R, f, offs, ii, jj;
  int emW0, emB0, emW1, emB1, emWo, emBo;
  int Wg[NB], iW[NB], ib_[NB], jW[NB], jb[NB], mW[NB], mb_[NB], uu[NB], outW[NB];
  int r1W[NB][3], r1b[NB][3], r2W[NB][3], r2b[NB][3];
  int a1W[NB][2], a1b[NB][2], a2W[NB][2], a2b[NB][2];
  int oW1[NB], ob1[NB], oW2[NB], ob2[NB];
};

static void build_map(PIdx& M, bool sorted) {
  if (!sorted) {
    M.R = 0; M.f = 1; M.offs = 2; M.ii = 174; M.jj = 175;
    M.emW0 = 3; M.emB0 = 4; M.emW1 = 5; M.emB1 = 6; M.emWo = 7; M.emBo = 8;
    for (int b = 0; b < NB; ++b) {
      int s = 9 + 33 * b;
      M.Wg[b] = s + 0;
      M.iW[b] = s + 1;  M.ib_[b] = s + 2;
      M.jW[b] = s + 3;  M.jb[b] = s + 4;
      for (int r = 0; r < 3; ++r) {
        M.r1W[b][r] = s + 5 + 4 * r; M.r1b[b][r] = s + 6 + 4 * r;
        M.r2W[b][r] = s + 7 + 4 * r; M.r2b[b][r] = s + 8 + 4 * r;
      }
      M.mW[b] = s + 17; M.mb_[b] = s + 18; M.uu[b] = s + 19;
      for (int r = 0; r < 2; ++r) {
        M.a1W[b][r] = s + 20 + 4 * r; M.a1b[b][r] = s + 21 + 4 * r;
        M.a2W[b][r] = s + 22 + 4 * r; M.a2b[b][r] = s + 23 + 4 * r;
      }
      M.oW1[b] = s + 28; M.ob1[b] = s + 29; M.oW2[b] = s + 30; M.ob2[b] = s + 31;
      M.outW[b] = s + 32;
    }
  } else {
    M.R = 0; M.f = 1; M.ii = 2; M.jj = 3; M.offs = 4;
    for (int b = 0; b < NB; ++b) {
      int s = 5 + 33 * b;           // 'blocks' < 'em'; block: 'int' < 'out'
      M.Wg[b] = s + 0;              // 'Wg' first (uppercase)
      for (int r = 0; r < 2; ++r) { // 'ares'
        M.a1W[b][r] = s + 1 + 4 * r; M.a1b[b][r] = s + 2 + 4 * r;
        M.a2W[b][r] = s + 3 + 4 * r; M.a2b[b][r] = s + 4 + 4 * r;
      }
      M.iW[b] = s + 9;  M.ib_[b] = s + 10;
      M.jW[b] = s + 11; M.jb[b] = s + 12;
      M.mW[b] = s + 13; M.mb_[b] = s + 14;
      for (int r = 0; r < 3; ++r) {
        M.r1W[b][r] = s + 15 + 4 * r; M.r1b[b][r] = s + 16 + 4 * r;
        M.r2W[b][r] = s + 17 + 4 * r; M.r2b[b][r] = s + 18 + 4 * r;
      }
      M.uu[b] = s + 27;
      M.outW[b] = s + 28;           // out: 'W' < 'res'
      M.oW1[b] = s + 29; M.ob1[b] = s + 30; M.oW2[b] = s + 31; M.ob2[b] = s + 32;
    }
    M.emW0 = 170; M.emB0 = 171; M.emW1 = 172; M.emB1 = 173; M.emWo = 174; M.emBo = 175;
  }
}

extern "C" void kernel_launch(void* const* d_in, const int* in_sizes, int n_in,
                              void* d_out, int out_size, void* d_ws, size_t ws_size,
                              hipStream_t stream) {
  (void)out_size; (void)ws_size;
  PIdx M;
  bool sorted = (n_in > 4 && in_sizes[2] == NE);  // idx_i in slot 2 -> sorted-key order
  build_map(M, sorted);

  const float* R    = (const float*)d_in[M.R];
  const float* fin  = (const float*)d_in[M.f];
  const float* offs = (const float*)d_in[M.offs];
  const int* idx_i  = (const int*)d_in[M.ii];
  const int* idx_j  = (const int*)d_in[M.jj];
  auto P = [&](int i) -> const float* { return (const float*)d_in[i]; };

  // ---- workspace carve ----
  char* cur = (char*)d_ws;
  auto alloc = [&](size_t bytes) -> char* {
    char* p = cur;
    cur += (bytes + 255) & ~(size_t)255;
    return p;
  };
  float*  x     = (float*)alloc((size_t)NA * FDIM * 4);
  float*  m     = (float*)alloc((size_t)NA * FDIM * 4);
  float*  obuf  = (float*)alloc((size_t)NA * FDIM * 4);
  __bf16* xa_bf = (__bf16*)alloc((size_t)NA * FDIM * 2);
  __bf16* t_bf  = (__bf16*)alloc((size_t)NA * FDIM * 2);
  __bf16* h_bf  = (__bf16*)alloc((size_t)NA * FDIM * 2);
  __bf16* fpad  = (__bf16*)alloc((size_t)NA * 32 * 2);
  __bf16* rbfb  = (__bf16*)alloc((size_t)NE * KRBF * 2 + 4096);  // +TDM overrun slack
  float*  prevO = (float*)alloc((size_t)NA * NOUTC * 4);
  float*  nh    = (float*)alloc(256);
  __bf16* pcur  = (__bf16*)cur;  // packed-weight region uses the rest

  auto packW = [&](int idx, int K, int N) -> const __bf16* {
    int Kp = ((K + 31) / 32) * 32, Np = ((N + 15) / 16) * 16;
    __bf16* dst = pcur;
    pcur += (size_t)Kp * Np;
    int tot = Kp * Np;
    k_pack<<<cdiv_i(tot, 256), 256, 0, stream>>>((const float*)d_in[idx], dst, K, N,
                                                 Np / 16, tot);
    return dst;
  };

  // ---- pack all GEMM B-operands ----
  const __bf16* emW0p = packW(M.emW0, 8, FDIM);
  const __bf16* emW1p = packW(M.emW1, FDIM, FDIM);
  const __bf16* emWop = packW(M.emWo, FDIM, FDIM);
  const __bf16 *WgP[NB], *iWP[NB], *jWP[NB], *mWP[NB], *outWP[NB];
  const __bf16 *r1P[NB][3], *r2P[NB][3], *a1P[NB][2], *a2P[NB][2], *o1P[NB], *o2P[NB];
  for (int b = 0; b < NB; ++b) {
    WgP[b] = packW(M.Wg[b], KRBF, FDIM);
    iWP[b] = packW(M.iW[b], FDIM, FDIM);
    jWP[b] = packW(M.jW[b], FDIM, FDIM);
    for (int r = 0; r < 3; ++r) {
      r1P[b][r] = packW(M.r1W[b][r], FDIM, FDIM);
      r2P[b][r] = packW(M.r2W[b][r], FDIM, FDIM);
    }
    mWP[b] = packW(M.mW[b], FDIM, FDIM);
    for (int r = 0; r < 2; ++r) {
      a1P[b][r] = packW(M.a1W[b][r], FDIM, FDIM);
      a2P[b][r] = packW(M.a2W[b][r], FDIM, FDIM);
    }
    o1P[b] = packW(M.oW1[b], FDIM, FDIM);
    o2P[b] = packW(M.oW2[b], FDIM, FDIM);
    outWP[b] = packW(M.outW[b], FDIM, NOUTC);  // padded to 2016 cols
  }

  // ---- init ----
  k_zero<<<cdiv_i(NTOT, 256), 256, 0, stream>>>((float*)d_out, NTOT);
  k_zero<<<1, 256, 0, stream>>>(nh, 1);

  // ---- RBF on edges ----
  k_rbf<<<cdiv_i(NE * KRBF, 256), 256, 0, stream>>>(R, offs, idx_i, idx_j, rbfb);

  // ---- elemental-modes embedding MLP ----
  k_pad_f<<<cdiv_i(NA * 32, 256), 256, 0, stream>>>(fin, fpad);
  launch_gemm<EP_SSP>(stream, MT_A, fpad, 32, 1, emW0p, 8, P(M.emB0),
                      nullptr, t_bf, FDIM, FDIM, nullptr, nullptr, nullptr, nullptr, 0);
  launch_gemm<EP_SSP>(stream, MT_A, t_bf, FDIM, 4, emW1p, 8, P(M.emB1),
                      nullptr, h_bf, FDIM, FDIM, nullptr, nullptr, nullptr, nullptr, 0);
  launch_gemm<EP_STORE>(stream, MT_A, h_bf, FDIM, 4, emWop, 8, P(M.emBo),
                        x, nullptr, FDIM, FDIM, nullptr, nullptr, nullptr, nullptr, 0);

  const int NEW = NA * FDIM;
  auto sspbf = [&](const float* src, __bf16* dst) {
    k_ssp_bf16<<<cdiv_i(NEW, 256), 256, 0, stream>>>(src, dst, NEW);
  };
  // dst = src + l2(ssp(l1(ssp(src))))
  auto residual = [&](float* dst, const float* src, const __bf16* W1p, const float* b1,
                      const __bf16* W2p, const float* b2) {
    sspbf(src, t_bf);
    launch_gemm<EP_SSP>(stream, MT_A, t_bf, FDIM, 4, W1p, 8, b1,
                        nullptr, h_bf, FDIM, FDIM, nullptr, nullptr, nullptr, nullptr, 0);
    launch_gemm<EP_ADD>(stream, MT_A, h_bf, FDIM, 4, W2p, 8, b2,
                        dst, nullptr, FDIM, FDIM, src, nullptr, nullptr, nullptr, 0);
  };

  // ---- interaction blocks ----
  for (int b = 0; b < NB; ++b) {
    // xa = ssp(x) (bf16 for GEMM + edge gather)
    sspbf(x, xa_bf);
    // m = ssp(xa @ Wi + bi)   (xi term)
    launch_gemm<EP_SSP>(stream, MT_A, xa_bf, FDIM, 4, iWP[b], 8, P(M.ib_[b]),
                        m, nullptr, FDIM, FDIM, nullptr, nullptr, nullptr, nullptr, 0);
    // m += segment_sum( (rbf@Wg) * ssp(xa[idx_j]@Wj + bj) )
    k_edge<<<NE / 16, 256, 0, stream>>>(rbfb, xa_bf, idx_i, idx_j, WgP[b], jWP[b],
                                        P(M.jb[b]), m);
    // interaction residual stack
    for (int r = 0; r < 3; ++r)
      residual(m, m, r1P[b][r], P(M.r1b[b][r]), r2P[b][r], P(M.r2b[b][r]));
    // x = u * x + ssp(m) @ Wm + bm
    sspbf(m, t_bf);
    launch_gemm<EP_UX>(stream, MT_A, t_bf, FDIM, 4, mWP[b], 8, P(M.mb_[b]),
                       x, nullptr, FDIM, FDIM, x, P(M.uu[b]), nullptr, nullptr, 0);
    // atomic residual stack
    for (int r = 0; r < 2; ++r)
      residual(x, x, a1P[b][r], P(M.a1b[b][r]), a2P[b][r], P(M.a2b[b][r]));
    // output residual: obuf = x + l2(ssp(l1(ssp(x))))
    residual(obuf, x, o1P[b], P(M.ob1[b]), o2P[b], P(M.ob2[b]));
    // o = ssp(obuf) @ Wout ; outputs += o ; nhloss term vs previous block's o
    sspbf(obuf, t_bf);
    launch_gemm<EP_OUT>(stream, MT_A, t_bf, FDIM, 4, outWP[b], 126, nullptr,
                        (float*)d_out, nullptr, NOUTC, NOUTC, nullptr, nullptr,
                        nh, prevO, (b > 0) ? 1 : 0);
  }

  // relu(outputs) + nhloss scalar (mean over N*NOUT per block, summed)
  k_finalize<<<cdiv_i(NTOT, 256), 256, 0, stream>>>((float*)d_out, nh);
}
